// GCN_V3_7301444403556
// MI455X (gfx1250) — compile-verified, matrix operand order checked by reading, not run on hardware
//
#include <hip/hip_runtime.h>
#include <math.h>

typedef __attribute__((ext_vector_type(16))) _Float16 v16h;
typedef __attribute__((ext_vector_type(8)))  _Float16 v8h;
typedef __attribute__((ext_vector_type(8)))  float    v8f;

#define NNODES 100000
#define NEDGES 500000
#define LDS_STRIDE 40   // halves per row: 80 B, keeps 16B segment alignment

// ---------------------------------------------------------------------------
// CDNA5 async-copy helpers (ASYNCcnt-tracked global->LDS DMA)
// ---------------------------------------------------------------------------
__device__ __forceinline__ unsigned lds_off(const void* p) {
    return (unsigned)(unsigned long long)(size_t)p;   // low 32 bits = LDS offset
}
__device__ __forceinline__ void async_b128(unsigned dst, const void* src) {
    asm volatile("global_load_async_to_lds_b128 %0, %1, off"
                 :: "v"(dst), "v"(src) : "memory");
}
__device__ __forceinline__ void wait_async0() {
    asm volatile("s_wait_asynccnt 0x0" ::: "memory");
}

// ---------------------------------------------------------------------------
// Degree / normalization
// ---------------------------------------------------------------------------
__global__ void gcn_deg_init(float* __restrict__ deg, int n) {
    int i = blockIdx.x * blockDim.x + threadIdx.x;
    if (i < n) deg[i] = 1.0f;                       // self-loop
}
__global__ void gcn_deg_count(const long long* __restrict__ dst,
                              float* __restrict__ deg, int e) {
    int i = blockIdx.x * blockDim.x + threadIdx.x;
    if (i < e) atomicAdd(&deg[(int)dst[i]], 1.0f);
}
__global__ void gcn_deg_rsqrt(float* __restrict__ deg, int n) {
    int i = blockIdx.x * blockDim.x + threadIdx.x;
    if (i < n) deg[i] = rsqrtf(deg[i]);
}

// ---------------------------------------------------------------------------
// One-time conversions (per layer)
// ---------------------------------------------------------------------------
__global__ void gcn_f32_to_f16(const float* __restrict__ in,
                               _Float16* __restrict__ out, long long total) {
    long long i = (long long)blockIdx.x * blockDim.x + threadIdx.x;
    if (i < total) out[i] = (_Float16)in[i];
}
// Wt[n][k] = f16(W[k][n])   (transpose so B-tile rows are contiguous in K)
__global__ void gcn_wt_cvt(const float* __restrict__ W,
                           _Float16* __restrict__ Wt, int K, int F) {
    int i = blockIdx.x * blockDim.x + threadIdx.x;
    if (i < K * F) {
        int k = i / F, n = i % F;
        Wt[(size_t)n * K + k] = (_Float16)W[i];
    }
}

// ---------------------------------------------------------------------------
// WMMA GEMM:  C[M,F] = A[M,K](f16) @ Wt[F,K](f16, pre-transposed)
// Block: 128x128 C tile, 8 wave32s (4 row-slabs x 2 col-slabs),
// each wave: 32x64 = 2x4 v_wmma_f32_16x16x32_f16 accumulators.
// Double-buffered LDS staged via global_load_async_to_lds_b128.
// K, F compile-time -> immediate addressing, known trip counts.
// ---------------------------------------------------------------------------
template <int K, int F>
__launch_bounds__(256)
__global__ void gcn_gemm_wmma(const _Float16* __restrict__ A,
                              const _Float16* __restrict__ Wt,
                              float* __restrict__ C, int M) {
    __shared__ _Float16 As[2][128 * LDS_STRIDE];
    __shared__ _Float16 Bs[2][128 * LDS_STRIDE];

    const int t    = threadIdx.x;
    const int lane = t & 31;
    const int wid  = t >> 5;
    const int wm   = wid & 3;        // 32-row slab
    const int wn   = wid >> 2;       // 64-col slab
    const int g    = lane >> 4;      // K-half selector
    const int lr   = lane & 15;
    const int m0   = blockIdx.x * 128;
    const int n0   = blockIdx.y * 128;
    constexpr int KT = K / 32;

    // staging coordinates: 512 16B chunks per tile, 2 per thread
    int  rr[2], cc[2]; bool avalid[2];
    #pragma unroll
    for (int p = 0; p < 2; ++p) {
        int idx = p * 256 + t;
        rr[p] = idx >> 2;            // 0..127
        cc[p] = (idx & 3) * 8;       // 0,8,16,24 halves
        avalid[p] = (m0 + rr[p]) < M;
    }

    auto stage = [&](int buf, int kt) {
        const int k0 = kt * 32;
        #pragma unroll
        for (int p = 0; p < 2; ++p) {
            if (avalid[p])
                async_b128(lds_off(&As[buf][rr[p] * LDS_STRIDE + cc[p]]),
                           A + (size_t)(m0 + rr[p]) * K + k0 + cc[p]);
            async_b128(lds_off(&Bs[buf][rr[p] * LDS_STRIDE + cc[p]]),
                       Wt + (size_t)(n0 + rr[p]) * K + k0 + cc[p]);
        }
    };

    v8f acc[2][4] = {};

    stage(0, 0);
    wait_async0();
    __syncthreads();

    for (int kt = 0; kt < KT; ++kt) {
        const int cur = kt & 1;
        if (kt + 1 < KT) stage(cur ^ 1, kt + 1);

        // fragment loads: ISA 16-bit 16x32 layout -> two contiguous 16B
        // segments per lane (lanes 0-15: K 0-7/16-23; lanes 16-31: K 8-15/24-31)
        v16h afr[2], bfr[4];
        #pragma unroll
        for (int i = 0; i < 2; ++i) {
            const _Float16* ap = &As[cur][(wm * 32 + i * 16 + lr) * LDS_STRIDE];
            v8h lo = *(const v8h*)(ap + 8 * g);
            v8h hi = *(const v8h*)(ap + 8 * g + 16);
            #pragma unroll
            for (int q = 0; q < 8; ++q) { afr[i][q] = lo[q]; afr[i][q + 8] = hi[q]; }
        }
        #pragma unroll
        for (int j = 0; j < 4; ++j) {
            const _Float16* bp = &Bs[cur][(wn * 64 + j * 16 + lr) * LDS_STRIDE];
            v8h lo = *(const v8h*)(bp + 8 * g);
            v8h hi = *(const v8h*)(bp + 8 * g + 16);
            #pragma unroll
            for (int q = 0; q < 8; ++q) { bfr[j][q] = lo[q]; bfr[j][q + 8] = hi[q]; }
        }
        #pragma unroll
        for (int i = 0; i < 2; ++i)
            #pragma unroll
            for (int j = 0; j < 4; ++j)
                acc[i][j] = __builtin_amdgcn_wmma_f32_16x16x32_f16(
                    false, afr[i], false, bfr[j], (short)0, acc[i][j], false, false);

        if (kt + 1 < KT) wait_async0();
        __syncthreads();
    }

    // ---- epilogue: D VGPR v holds M = v + 8*g, N = lane%16 ----
    // fast path for full blocks (all but the last M-block): no bounds checks,
    // one base pointer per row-slab, constant column offsets.
    const int colb = n0 + wn * 64 + lr;
    if (m0 + 128 <= M) {
        #pragma unroll
        for (int i = 0; i < 2; ++i) {
            float* p = C + (size_t)(m0 + wm * 32 + i * 16 + g * 8) * F + colb;
            #pragma unroll
            for (int v = 0; v < 8; ++v) {
                #pragma unroll
                for (int j = 0; j < 4; ++j)
                    p[(size_t)v * F + j * 16] = acc[i][j][v];
            }
        }
    } else {
        #pragma unroll
        for (int i = 0; i < 2; ++i) {
            const int row0 = m0 + wm * 32 + i * 16 + g * 8;
            float* p = C + (size_t)row0 * F + colb;
            #pragma unroll
            for (int v = 0; v < 8; ++v) {
                if (row0 + v < M) {
                    #pragma unroll
                    for (int j = 0; j < 4; ++j)
                        p[(size_t)v * F + j * 16] = acc[i][j][v];
                }
            }
        }
    }
}

// ---------------------------------------------------------------------------
// Aggregation (2D row-based: no div/mod)
// ---------------------------------------------------------------------------
template <int F>
__global__ void gcn_selfloop_init(const float* __restrict__ hw,
                                  const float* __restrict__ dinv,
                                  float* __restrict__ agg) {
    const int row = blockIdx.x;
    const int f   = blockIdx.y * blockDim.x + threadIdx.x;
    const float w = dinv[row];
    const size_t idx = (size_t)row * F + f;
    agg[idx] = w * w * hw[idx];
}

// one wave32 per edge, 8 edges per block
template <int F>
__launch_bounds__(256)
__global__ void gcn_edge_scatter(const long long* __restrict__ src,
                                 const long long* __restrict__ dst,
                                 const float* __restrict__ dinv,
                                 const float* __restrict__ hw,
                                 float* __restrict__ agg, int E) {
    const int e = blockIdx.x * 8 + (threadIdx.x >> 5);
    if (e >= E) return;
    const int lane = threadIdx.x & 31;
    const int s = (int)src[e];
    const int d = (int)dst[e];
    const float w = dinv[s] * dinv[d];
    const float* in  = hw  + (size_t)s * F;
    float*       out = agg + (size_t)d * F;
    #pragma unroll
    for (int f = lane; f < F; f += 32)
        atomicAdd(&out[f], w * in[f]);
}

template <int F>
__global__ void gcn_bias_gelu_f16(const float* __restrict__ agg,
                                  const float* __restrict__ bias,
                                  _Float16* __restrict__ out) {
    const int row = blockIdx.x;
    const int f   = blockIdx.y * blockDim.x + threadIdx.x;
    const size_t idx = (size_t)row * F + f;
    float x = agg[idx] + bias[f];
    x = 0.5f * x * (1.0f + erff(x * 0.70710678118654752f));
    out[idx] = (_Float16)x;
}

template <int F>
__global__ void gcn_bias_f32(const float* __restrict__ agg,
                             const float* __restrict__ bias,
                             float* __restrict__ out) {
    const int row = blockIdx.x;
    const int f   = blockIdx.y * blockDim.x + threadIdx.x;
    const size_t idx = (size_t)row * F + f;
    out[idx] = agg[idx] + bias[f];
}

// ---------------------------------------------------------------------------
// Host-side orchestration
// ---------------------------------------------------------------------------
template <int K, int F, bool LAST>
static void run_layer(const float* Wf, const float* bf,
                      _Float16* hf, _Float16* wt, float* hw, float* agg,
                      const float* dinv, const long long* src, const long long* dst,
                      float* out_f32, hipStream_t stream) {
    gcn_wt_cvt<<<(K * F + 255) / 256, 256, 0, stream>>>(Wf, wt, K, F);

    dim3 ggrid((NNODES + 127) / 128, F / 128);
    gcn_gemm_wmma<K, F><<<ggrid, 256, 0, stream>>>(hf, wt, hw, NNODES);

    const int eblk  = (F < 256) ? F : 256;
    dim3 egrid(NNODES, F / eblk);
    gcn_selfloop_init<F><<<egrid, eblk, 0, stream>>>(hw, dinv, agg);
    gcn_edge_scatter<F><<<(NEDGES + 7) / 8, 256, 0, stream>>>(src, dst, dinv, hw, agg, NEDGES);
    if (LAST) gcn_bias_f32<F>     <<<egrid, eblk, 0, stream>>>(agg, bf, out_f32);
    else      gcn_bias_gelu_f16<F><<<egrid, eblk, 0, stream>>>(agg, bf, hf);
}

extern "C" void kernel_launch(void* const* d_in, const int* in_sizes, int n_in,
                              void* d_out, int out_size, void* d_ws, size_t ws_size,
                              hipStream_t stream) {
    const float*     x  = (const float*)d_in[0];
    const long long* ei = (const long long*)d_in[1];   // int64 per reference
    const float* W1 = (const float*)d_in[2]; const float* b1 = (const float*)d_in[3];
    const float* W2 = (const float*)d_in[4]; const float* b2 = (const float*)d_in[5];
    const float* W3 = (const float*)d_in[6]; const float* b3 = (const float*)d_in[7];
    const float* W4 = (const float*)d_in[8]; const float* b4 = (const float*)d_in[9];

    const long long* src = ei;            // edge_index[0]
    const long long* dst = ei + NEDGES;   // edge_index[1]

    const size_t MB  = (size_t)1 << 20;
    const size_t BIG = 308 * MB;          // >= N*768*4
    char* wsb = (char*)d_ws;
    float*    dinv = (float*)wsb;                                  // N f32
    float*    hw   = (float*)(wsb + MB);                           // GEMM out
    float*    agg  = (float*)(wsb + MB + BIG);                     // scatter acc
    _Float16* hf   = (_Float16*)(wsb + MB + 2 * BIG);              // f16 acts
    _Float16* wt   = (_Float16*)(wsb + MB + 2 * BIG + 156 * MB);   // f16 W^T

    // normalization (recomputed every call: deterministic)
    gcn_deg_init <<<(NNODES + 255) / 256, 256, 0, stream>>>(dinv, NNODES);
    gcn_deg_count<<<(NEDGES + 255) / 256, 256, 0, stream>>>(dst, dinv, NEDGES);
    gcn_deg_rsqrt<<<(NNODES + 255) / 256, 256, 0, stream>>>(dinv, NNODES);

    // x -> f16 activations
    {
        long long tot = (long long)NNODES * 128;
        gcn_f32_to_f16<<<(unsigned)((tot + 255) / 256), 256, 0, stream>>>(x, hf, tot);
    }

    run_layer<128, 512, false>(W1, b1, hf, wt, hw, agg, dinv, src, dst, nullptr, stream);
    run_layer<512, 768, false>(W2, b2, hf, wt, hw, agg, dinv, src, dst, nullptr, stream);
    run_layer<768, 512, false>(W3, b3, hf, wt, hw, agg, dinv, src, dst, nullptr, stream);
    run_layer<512, 128, true >(W4, b4, hf, wt, hw, agg, dinv, src, dst, (float*)d_out, stream);
}
